// ACMGCN_80865644249436
// MI455X (gfx1250) — matrix operand order repaired and checked by reading, not compiled, not run on hardware
//
#include <hip/hip_runtime.h>
#include <hip/hip_bf16.h>
#include <math.h>

// ---------------------------------------------------------------------------
// ACM-GCN on gfx1250 (MI455X).
//   GEMMs  : v_wmma_f32_16x16x32_f16 (f16 operands, f32 accumulate)
//   SpMM   : gather + global_atomic_add_f32 (all traffic resident in 192MB L2)
//   Epilog : wave-per-row attention combine (wave32 shfl_xor reductions)
// ---------------------------------------------------------------------------

typedef __attribute__((ext_vector_type(16))) _Float16 v16h;
typedef __attribute__((ext_vector_type(8)))  float    v8f;

// ---------------- degree / inverse-degree -----------------------------------

__global__ void deg_kernel(const int* __restrict__ rows, float* __restrict__ deg, int E) {
    int t = blockIdx.x * blockDim.x + threadIdx.x;
    if (t < E) atomicAdd(&deg[rows[t]], 1.0f);
}

__global__ void invdeg_kernel(float* __restrict__ deg, int N) {
    int t = blockIdx.x * blockDim.x + threadIdx.x;
    if (t < N) {
        float d = deg[t];
        deg[t] = (d > 0.0f) ? (1.0f / d) : 0.0f;
    }
}

// ---------------- weight packing into WMMA B-operand layout -----------------
// Packed index: ((kt*NTtot + nt)*32 + lane)*16 + j   (j = half index 0..15)
// B (32x16 f16) lane layout (from ISA 16-bit sparse B table, scaled):
//   lanes 0-15 : N = lane,    K = j (j<8), j+8 (j>=8)          (VGPR j/2, lo/hi)
//   lanes16-31 : N = lane-16, K = 8 + j (j<8), 16 + j (j>=8)
__global__ void pack_weights(const float* __restrict__ w0, const float* __restrict__ w1,
                             const float* __restrict__ w2, _Float16* __restrict__ packed,
                             int K, int Nout) {
    int KT = K >> 5, ntPer = Nout >> 4, NTtot = 3 * ntPer;
    int total = KT * NTtot * 512;
    int tid = blockIdx.x * blockDim.x + threadIdx.x;
    if (tid >= total) return;
    int j    = tid & 15;
    int t    = tid >> 4;
    int lane = t & 31;
    t >>= 5;
    int nt = t % NTtot;
    int kt = t / NTtot;
    int m   = nt / ntPer;
    int col = (nt % ntPer) * 16 + (lane & 15);
    int k   = kt * 32 + (j & 7) + ((j & 8) ? 16 : 0) + ((lane & 16) ? 8 : 0);
    const float* w = (m == 0) ? w0 : ((m == 1) ? w1 : w2);
    packed[tid] = (_Float16)w[(size_t)k * Nout + col];
}

// ---------------- fused triple GEMM: o0=h@W0, o1=h@W1, o2=relu(h@W2) --------
// One wave per 16-row M tile. KT/ntPer are compile-time so the a[] register
// array is statically indexed (no v_movrels/m0) and loops fully unroll.
// A (16x32 f16) lane layout (ISA 7.12.2):
//   lanes 0-15 : M = lane,    K = {0..7, 16..23}
//   lanes16-31 : M = lane-16, K = {8..15, 24..31}
template <int KT, int NTPER>
__global__ __launch_bounds__(256)
void gemm3_wmma(const float* __restrict__ h, const _Float16* __restrict__ pw,
                float* __restrict__ o0, float* __restrict__ o1, float* __restrict__ o2,
                int nrows) {
    constexpr int K     = KT * 32;
    constexpr int Nout  = NTPER * 16;
    constexpr int NTtot = 3 * NTPER;

    int wave = (blockIdx.x * blockDim.x + threadIdx.x) >> 5;
    int lane = threadIdx.x & 31;
    int mtiles = nrows >> 4;
    if (wave >= mtiles) return;

    int mbase    = wave << 4;
    int row      = mbase + (lane & 15);
    int koffBase = (lane & 16) ? 8 : 0;
    int rbase    = mbase + ((lane & 16) ? 8 : 0);

    v16h a[KT];
    #pragma unroll
    for (int kt = 0; kt < KT; ++kt) {
        const float* ap = h + (size_t)row * K + kt * 32 + koffBase;
        float4 p0 = *(const float4*)(ap);
        float4 p1 = *(const float4*)(ap + 4);
        float4 p2 = *(const float4*)(ap + 16);
        float4 p3 = *(const float4*)(ap + 20);
        v16h av;
        av[0]  = (_Float16)p0.x; av[1]  = (_Float16)p0.y;
        av[2]  = (_Float16)p0.z; av[3]  = (_Float16)p0.w;
        av[4]  = (_Float16)p1.x; av[5]  = (_Float16)p1.y;
        av[6]  = (_Float16)p1.z; av[7]  = (_Float16)p1.w;
        av[8]  = (_Float16)p2.x; av[9]  = (_Float16)p2.y;
        av[10] = (_Float16)p2.z; av[11] = (_Float16)p2.w;
        av[12] = (_Float16)p3.x; av[13] = (_Float16)p3.y;
        av[14] = (_Float16)p3.z; av[15] = (_Float16)p3.w;
        a[kt] = av;
    }

    #pragma unroll
    for (int m = 0; m < 3; ++m) {
        float* outp = (m == 0) ? o0 : ((m == 1) ? o1 : o2);
        #pragma unroll
        for (int ntl = 0; ntl < NTPER; ++ntl) {
            const int nt = m * NTPER + ntl;
            v8f c = {};
            #pragma unroll
            for (int kt = 0; kt < KT; ++kt) {
                v16h b = *(const v16h*)(pw + (((size_t)kt * NTtot + nt) * 32 + lane) * 16);
                c = __builtin_amdgcn_wmma_f32_16x16x32_f16(
                        /*neg_a=*/false, a[kt], /*neg_b=*/false, b,
                        /*c_mod=*/(short)0, c, /*reuse_a=*/false, /*reuse_b=*/false);
            }
            int coln = ntl * 16 + (lane & 15);
            #pragma unroll
            for (int v = 0; v < 8; ++v) {
                float val = c[v];
                if (m == 2) val = fmaxf(val, 0.0f);     // out_mlp relu fused
                outp[(size_t)(rbase + v) * Nout + coln] = val;
            }
        }
    }
}

// ---------------- dual SpMM scatter: acc[row] += src[col]  ------------------
// One wave per edge; handles both hl and hh (edge list read once).
template <int W>
__global__ __launch_bounds__(256)
void spmm_scatter(const int* __restrict__ rows, const int* __restrict__ cols,
                  const float* __restrict__ hl, const float* __restrict__ hh,
                  float* __restrict__ accL, float* __restrict__ accH, int E) {
    int e = (blockIdx.x * blockDim.x + threadIdx.x) >> 5;
    int lane = threadIdx.x & 31;
    if (e >= E) return;
    int r = rows[e], c = cols[e];
    constexpr int PER = W / 32;
    size_t sb = (size_t)c * W + lane * PER;
    size_t db = (size_t)r * W + lane * PER;
    float vl[PER], vh[PER];
    if constexpr (PER == 4) {
        float4 a = *(const float4*)(hl + sb);
        float4 b = *(const float4*)(hh + sb);
        vl[0] = a.x; vl[1] = a.y; vl[2] = a.z; vl[3] = a.w;
        vh[0] = b.x; vh[1] = b.y; vh[2] = b.z; vh[3] = b.w;
    } else {
        float2 a = *(const float2*)(hl + sb);
        float2 b = *(const float2*)(hh + sb);
        vl[0] = a.x; vl[1] = a.y;
        vh[0] = b.x; vh[1] = b.y;
    }
    #pragma unroll
    for (int i = 0; i < PER; ++i) {
        atomicAdd(accL + db + i, vl[i]);
        atomicAdd(accH + db + i, vh[i]);
    }
}

// ---------------- attention combine epilogue (wave per row) -----------------
// ol = relu(invd*accL); oh = relu(hh - invd*accH); om = mlp (pre-relu'd)
// logits -> sigmoid -> @att(3x3)/T -> softmax -> blend; optional output relu.
template <int W, bool RELU_OUT>
__global__ __launch_bounds__(256)
void combine_attn(const float* __restrict__ hh, const float* __restrict__ mlp,
                  const float* __restrict__ accL, const float* __restrict__ accH,
                  const float* __restrict__ invd,
                  const float* __restrict__ vlw, const float* __restrict__ vhw,
                  const float* __restrict__ vmw, const float* __restrict__ att,
                  float* __restrict__ out, int N) {
    int rowi = (blockIdx.x * blockDim.x + threadIdx.x) >> 5;
    int lane = threadIdx.x & 31;
    if (rowi >= N) return;
    constexpr int PER = W / 32;
    size_t base = (size_t)rowi * W + lane * PER;
    float id = invd[rowi];

    float ol[PER], oh[PER], om[PER];
    float sl = 0.0f, sh = 0.0f, sm = 0.0f;
    #pragma unroll
    for (int i = 0; i < PER; ++i) {
        int c = lane * PER + i;
        ol[i] = fmaxf(id * accL[base + i], 0.0f);
        oh[i] = fmaxf(hh[base + i] - id * accH[base + i], 0.0f);
        om[i] = mlp[base + i];
        sl += ol[i] * vlw[c];
        sh += oh[i] * vhw[c];
        sm += om[i] * vmw[c];
    }
    #pragma unroll
    for (int off = 16; off > 0; off >>= 1) {     // wave32 butterfly: all lanes get sum
        sl += __shfl_xor(sl, off, 32);
        sh += __shfl_xor(sh, off, 32);
        sm += __shfl_xor(sm, off, 32);
    }
    float gl = 1.0f / (1.0f + expf(-sl));
    float gh = 1.0f / (1.0f + expf(-sh));
    float gm = 1.0f / (1.0f + expf(-sm));
    float g[3];
    #pragma unroll
    for (int k = 0; k < 3; ++k)
        g[k] = (gl * att[0 * 3 + k] + gh * att[1 * 3 + k] + gm * att[2 * 3 + k]) * (1.0f / 3.0f);
    float mx = fmaxf(g[0], fmaxf(g[1], g[2]));
    float e0 = expf(g[0] - mx), e1 = expf(g[1] - mx), e2 = expf(g[2] - mx);
    float inv = 1.0f / (e0 + e1 + e2);
    float a0 = e0 * inv, a1 = e1 * inv, a2 = e2 * inv;

    #pragma unroll
    for (int i = 0; i < PER; ++i) {
        float v = 3.0f * (a0 * ol[i] + a1 * oh[i] + a2 * om[i]);
        if (RELU_OUT) v = fmaxf(v, 0.0f);
        out[base + i] = v;
    }
}

// ---------------------------------------------------------------------------

extern "C" void kernel_launch(void* const* d_in, const int* in_sizes, int n_in,
                              void* d_out, int out_size, void* d_ws, size_t ws_size,
                              hipStream_t stream) {
    const int F = 128, H = 128, C = 64;
    const float* x    = (const float*)d_in[0];
    const int*   ei   = (const int*)d_in[1];
    const float* Wl1  = (const float*)d_in[2];
    const float* Wh1  = (const float*)d_in[3];
    const float* Wm1  = (const float*)d_in[4];
    const float* vl1  = (const float*)d_in[5];
    const float* vh1  = (const float*)d_in[6];
    const float* vm1  = (const float*)d_in[7];
    const float* att1 = (const float*)d_in[8];
    const float* Wl2  = (const float*)d_in[9];
    const float* Wh2  = (const float*)d_in[10];
    const float* Wm2  = (const float*)d_in[11];
    const float* vl2  = (const float*)d_in[12];
    const float* vh2  = (const float*)d_in[13];
    const float* vm2  = (const float*)d_in[14];
    const float* att2 = (const float*)d_in[15];

    int N = in_sizes[0] / F;          // 100000
    int E = in_sizes[1] / 2;          // 1600000
    const int* rows = ei;
    const int* cols = ei + E;

    // workspace layout (floats)
    float* ws   = (float*)d_ws;
    size_t NB   = (size_t)N * H;
    size_t off0 = ((size_t)N + 255) & ~(size_t)255;
    float* invd = ws;
    float* B1 = ws + off0;            // hl1 -> fea
    float* B2 = B1 + NB;              // hh1 -> (hl2 | hh2)
    float* B3 = B2 + NB;              // mlp1 -> mlp2
    float* B4 = B3 + NB;              // accL
    float* B5 = B4 + NB;              // accH
    _Float16* pk1 = (_Float16*)(B5 + NB);   // 4*24*512 halves
    _Float16* pk2 = pk1 + 4 * 24 * 512;     // 4*12*512 halves

    int mtiles     = N / 16;
    int gemmBlocks = (mtiles * 32 + 255) / 256;
    int edgeBlocks = (E * 32 + 255) / 256;   // wave per edge
    int rowBlocks  = (N * 32 + 255) / 256;   // wave per row

    // ---- degree -> inv_deg
    hipMemsetAsync(invd, 0, (size_t)N * sizeof(float), stream);
    deg_kernel<<<(E + 255) / 256, 256, 0, stream>>>(rows, invd, E);
    invdeg_kernel<<<(N + 255) / 256, 256, 0, stream>>>(invd, N);

    // ---- layer 1
    pack_weights<<<(4 * 24 * 512 + 255) / 256, 256, 0, stream>>>(Wl1, Wh1, Wm1, pk1, F, H);
    gemm3_wmma<4, 8><<<gemmBlocks, 256, 0, stream>>>(x, pk1, B1, B2, B3, N);
    hipMemsetAsync(B4, 0, NB * sizeof(float), stream);
    hipMemsetAsync(B5, 0, NB * sizeof(float), stream);
    spmm_scatter<128><<<edgeBlocks, 256, 0, stream>>>(rows, cols, B1, B2, B4, B5, E);
    combine_attn<128, true><<<rowBlocks, 256, 0, stream>>>(B2, B3, B4, B5, invd,
                                                           vl1, vh1, vm1, att1, B1, N);

    // ---- layer 2 (input fea = B1)
    pack_weights<<<(4 * 12 * 512 + 255) / 256, 256, 0, stream>>>(Wl2, Wh2, Wm2, pk2, H, C);
    float* hl2  = B2;
    float* hh2  = B2 + (size_t)N * C;
    float* mlp2 = B3;
    gemm3_wmma<4, 4><<<gemmBlocks, 256, 0, stream>>>(B1, pk2, hl2, hh2, mlp2, N);
    hipMemsetAsync(B4, 0, (size_t)N * C * sizeof(float), stream);
    hipMemsetAsync(B5, 0, (size_t)N * C * sizeof(float), stream);
    spmm_scatter<64><<<edgeBlocks, 256, 0, stream>>>(rows, cols, hl2, hh2, B4, B5, E);
    combine_attn<64, false><<<rowBlocks, 256, 0, stream>>>(hh2, mlp2, B4, B5, invd,
                                                           vl2, vh2, vm2, att2,
                                                           (float*)d_out, N);
}
